// NeighborPruningModel_25872882991240
// MI455X (gfx1250) — matrix operand order repaired and checked by reading, not compiled, not run on hardware
//
#include <hip/hip_runtime.h>

// ---------------- problem constants (from reference) ----------------
#define NNODES 100000
#define NEDGES 800000
#define NB     1024
#define NK     10
#define HD     512
#define MROWS  (NB * NK)   // 10240
#define FCIN   (3 * HD)    // 1536
#define FCH    256
#define BN_EPS 1e-5f

typedef __attribute__((ext_vector_type(16))) __bf16 v16bf;
typedef __attribute__((ext_vector_type(4)))  __bf16 v4bf;
typedef __attribute__((ext_vector_type(8)))  float  v8f;

// ---------------- h0 = node_h @ w_init + b_init  [N,20]x[20,512] ----------------
__global__ void k_init_gemm(const float* __restrict__ nh, const float* __restrict__ w,
                            const float* __restrict__ bias, float* __restrict__ h) {
    __shared__ float row[20];
    int n = blockIdx.x;
    int c = threadIdx.x;                  // 512 threads
    if (c < 20) row[c] = nh[(size_t)n * 20 + c];
    __syncthreads();
    float acc = bias[c];
#pragma unroll
    for (int k = 0; k < 20; ++k) acc += row[k] * w[k * HD + c];
    h[(size_t)n * HD + c] = acc;
}

// ---------------- degree: deg[dst] += 1 ----------------
__global__ void k_deg(const int* __restrict__ dst, float* __restrict__ deg) {
    int e = blockIdx.x * blockDim.x + threadIdx.x;
    if (e < NEDGES) atomicAdd(&deg[dst[e]], 1.0f);
}

// ---------------- edge scatter: msum[dst] += h[src], one wave per edge ----------------
__global__ void k_edge_scatter(const float* __restrict__ h, const int* __restrict__ src,
                               const int* __restrict__ dst, float* __restrict__ msum) {
    int wid  = (blockIdx.x * blockDim.x + threadIdx.x) >> 5;   // global wave id = edge
    int lane = threadIdx.x & 31;
    if (wid >= NEDGES) return;
    int s = src[wid], d = dst[wid];
    const float* hs = h + (size_t)s * HD;
    float*       md = msum + (size_t)d * HD;
#pragma unroll
    for (int j = 0; j < HD / 32; ++j)
        atomicAdd(&md[lane + j * 32], hs[lane + j * 32]);
}

// ---------------- GIN combine (in-place into msum) + column sum/sumsq ----------------
#define GIN_RPB 128
__global__ void k_gin_combine(const float* __restrict__ h, float* __restrict__ msum,
                              const float* __restrict__ deg,
                              float* __restrict__ csum, float* __restrict__ css) {
    int c  = threadIdx.x;                 // 512 threads = columns
    int r0 = blockIdx.x * GIN_RPB;
    int r1 = min(r0 + GIN_RPB, NNODES);
    float s = 0.f, ss = 0.f;
    for (int r = r0; r < r1; ++r) {
        size_t idx = (size_t)r * HD + c;
        float x = h[idx] + msum[idx] / fmaxf(deg[r], 1.0f);
        msum[idx] = x;
        s += x; ss += x * x;
    }
    atomicAdd(&csum[c], s);
    atomicAdd(&css[c], ss);
}

// ---------------- column stats for FC preactivations ----------------
__global__ void k_col_stats(const float* __restrict__ x, float* __restrict__ csum,
                            float* __restrict__ css, int rows, int rpb) {
    int c = threadIdx.x;                  // blockDim.x == cols
    int cols = blockDim.x;
    int r0 = blockIdx.x * rpb;
    int r1 = min(r0 + rpb, rows);
    float s = 0.f, ss = 0.f;
    for (int r = r0; r < r1; ++r) {
        float v = x[(size_t)r * cols + c];
        s += v; ss += v * v;
    }
    atomicAdd(&csum[c], s);
    atomicAdd(&css[c], ss);
}

// ---------------- BN (batch stats) + ReLU, elementwise ----------------
__global__ void k_bn_relu(const float* __restrict__ x, const float* __restrict__ csum,
                          const float* __restrict__ css, const float* __restrict__ g,
                          const float* __restrict__ be, float* __restrict__ y,
                          int rows, int cols) {
    size_t i = (size_t)blockIdx.x * blockDim.x + threadIdx.x;
    size_t total = (size_t)rows * cols;
    if (i >= total) return;
    int c = (int)(i % cols);
    float inv = 1.0f / (float)rows;
    float m = csum[c] * inv;
    float v = css[c] * inv - m * m;
    float rs = rsqrtf(v + BN_EPS);
    float val = (x[i] - m) * rs * g[c] + be[c];
    y[i] = fmaxf(val, 0.f);
}

// ---------------- per-graph pooling, one wave per node ----------------
__global__ void k_pool(const float* __restrict__ h, const int* __restrict__ ng,
                       float* __restrict__ gsum, float* __restrict__ gcnt) {
    int n    = (blockIdx.x * blockDim.x + threadIdx.x) >> 5;
    int lane = threadIdx.x & 31;
    if (n >= NNODES) return;
    int g = ng[n];
    const float* hr = h + (size_t)n * HD;
    float*       gr = gsum + (size_t)g * HD;
#pragma unroll
    for (int j = 0; j < HD / 32; ++j)
        atomicAdd(&gr[lane + j * 32], hr[lane + j * 32]);
    if (lane == 0) atomicAdd(&gcnt[g], 1.0f);
}

// ---------------- build X = [qemb | pg_emb | neigh_emb]  [10240, 1536] ----------------
__global__ void k_concat(const float* __restrict__ gsum, const float* __restrict__ gcnt,
                         const float* __restrict__ pg, const float* __restrict__ ne,
                         float* __restrict__ X) {
    size_t i = (size_t)blockIdx.x * blockDim.x + threadIdx.x;
    if (i >= (size_t)MROWS * FCIN) return;
    int row = (int)(i / FCIN);
    int col = (int)(i % FCIN);
    int b = row / NK;
    float v;
    if (col < HD)            v = gsum[(size_t)b * HD + col] / fmaxf(gcnt[b], 1.0f);
    else if (col < 2 * HD)   v = pg[(size_t)b * HD + (col - HD)];
    else                     v = ne[(size_t)row * HD + (col - 2 * HD)];
    X[i] = v;
}

// ---------------- WMMA bf16 GEMM: Y[M,Nc] = X[M,Kd] @ W[Kd,Nc] + bias ----------------
// block = 128 threads (4 waves); each wave owns one 16x16 C tile; block tile = 64(M) x 16(N)
// Tiles are staged into LDS *pre-swizzled into WMMA fragment order*, so each lane's
// 16-element bf16 fragment is one contiguous, 8B/32B-aligned chunk -> ds_load_b128.
__global__ void k_gemm_bias_wmma(const float* __restrict__ X, const float* __restrict__ W,
                                 const float* __restrict__ bias, float* __restrict__ Y,
                                 int M, int Kd, int Nc) {
    __shared__ v16bf Afrag[4][32];   // [wave][lane] -> 16 contiguous bf16 (4096 B)
    __shared__ v16bf Bfrag[32];      // [lane]       -> 16 contiguous bf16 (1024 B)
    __bf16* Af = (__bf16*)Afrag;
    __bf16* Bf = (__bf16*)Bfrag;

    int tid  = threadIdx.x;
    int wv   = tid >> 5;
    int lane = tid & 31;
    int half = lane >> 4;
    int mr   = lane & 15;
    int mBase = blockIdx.x * 64;
    int nBase = blockIdx.y * 16;

    v8f acc = {};

    for (int k0 = 0; k0 < Kd; k0 += 32) {
        // ---- stage A (64x32 f32 -> bf16 fragments): 512 float4 loads / 128 thr ----
#pragma unroll
        for (int jq = 0; jq < 4; ++jq) {
            int q  = tid + 128 * jq;          // quad index 0..511
            int eb4 = q * 4;                  // element base 0..2044
            int r  = eb4 >> 5;                // tile row 0..63
            int cb = eb4 & 31;                // k within tile, multiple of 4
            const float4 xv = *(const float4*)(X + (size_t)(mBase + r) * Kd + (k0 + cb));
            // inverse of A-fragment map: k -> (half, e); cb quad stays in one 8-group
            int fh, fe;
            if (cb < 16) { fh = cb >> 3;        fe = cb & 7; }
            else         { fh = (cb - 16) >> 3; fe = 8 + ((cb - 16) & 7); }
            int slot = (((r >> 4) * 32) + ((r & 15) + 16 * fh)) * 16 + fe; // 8B aligned
            v4bf p;
            p[0] = (__bf16)xv.x; p[1] = (__bf16)xv.y;
            p[2] = (__bf16)xv.z; p[3] = (__bf16)xv.w;
            *(v4bf*)(Af + slot) = p;
        }
        // ---- stage B (32x16): one float4 per thread, coalesced along N ----
        {
            int eb4 = tid * 4;                // 0..508
            int kk  = eb4 >> 4;               // k row 0..31
            int nb  = eb4 & 15;               // n base, multiple of 4
            const float4 wv4 = *(const float4*)(W + (size_t)(k0 + kk) * Nc + (nBase + nb));
            int fh = (kk & 15) >> 3, fe = (kk & 7) + ((kk >> 4) << 3);   // inverse of B-fragment map
            Bf[((nb + 0) + 16 * fh) * 16 + fe] = (__bf16)wv4.x;
            Bf[((nb + 1) + 16 * fh) * 16 + fe] = (__bf16)wv4.y;
            Bf[((nb + 2) + 16 * fh) * 16 + fe] = (__bf16)wv4.z;
            Bf[((nb + 3) + 16 * fh) * 16 + fe] = (__bf16)wv4.w;
        }
        __syncthreads();

        // fragment loads: contiguous 32B -> 2x ds_load_b128 each
        v16bf af = Afrag[wv][lane];
        v16bf bf = Bfrag[lane];
        acc = __builtin_amdgcn_wmma_f32_16x16x32_bf16(
            false, af, false, bf, (short)0, acc, false, false);
        __syncthreads();
    }

    // C/D layout: VGPR r -> M = r + 8*half ; N = lane&15
#pragma unroll
    for (int r = 0; r < 8; ++r) {
        int row = mBase + wv * 16 + half * 8 + r;
        int col = nBase + mr;
        Y[(size_t)row * Nc + col] = acc[r] + bias[col];
    }
}

// ---------------- preds = sigmoid(H3 @ w_fc4 + b_fc4), wave per row ----------------
__global__ void k_fc4_sigmoid(const float* __restrict__ H3, const float* __restrict__ w,
                              const float* __restrict__ b, float* __restrict__ preds) {
    int row  = (blockIdx.x * blockDim.x + threadIdx.x) >> 5;
    int lane = threadIdx.x & 31;
    if (row >= MROWS) return;
    const float* hr = H3 + (size_t)row * FCH;
    float acc = 0.f;
#pragma unroll
    for (int j = 0; j < FCH / 32; ++j)
        acc += hr[lane + 32 * j] * w[lane + 32 * j];
#pragma unroll
    for (int off = 16; off > 0; off >>= 1)
        acc += __shfl_xor(acc, off, 32);
    if (lane == 0) preds[row] = 1.0f / (1.0f + __expf(-(acc + b[0])));
}

// =====================================================================
extern "C" void kernel_launch(void* const* d_in, const int* in_sizes, int n_in,
                              void* d_out, int out_size, void* d_ws, size_t ws_size,
                              hipStream_t stream) {
    const float* node_h   = (const float*)d_in[0];
    const float* pg_emb   = (const float*)d_in[1];
    const float* neigh    = (const float*)d_in[2];
    // d_in[3] = class_w (unused by reference)
    const int*   src      = (const int*)d_in[4];
    const int*   dst      = (const int*)d_in[5];
    const int*   ngraph   = (const int*)d_in[6];
    const float* w_init   = (const float*)d_in[7];
    const float* b_init   = (const float*)d_in[8];
    const float* g1       = (const float*)d_in[9];
    const float* be1      = (const float*)d_in[10];
    const float* g2       = (const float*)d_in[11];
    const float* be2      = (const float*)d_in[12];
    const float* w_fc     = (const float*)d_in[13];
    const float* b_fc     = (const float*)d_in[14];
    const float* w_fc2    = (const float*)d_in[15];
    const float* b_fc2    = (const float*)d_in[16];
    const float* w_fc3    = (const float*)d_in[17];
    const float* b_fc3    = (const float*)d_in[18];
    const float* w_fc4    = (const float*)d_in[19];
    const float* b_fc4    = (const float*)d_in[20];
    const float* gb       = (const float*)d_in[21];
    const float* bb       = (const float*)d_in[22];
    const float* gb2      = (const float*)d_in[23];
    const float* bb2      = (const float*)d_in[24];
    const float* gb3      = (const float*)d_in[25];
    const float* bb3      = (const float*)d_in[26];

    // ---------------- workspace carving ----------------
    const size_t BIG = (size_t)NNODES * HD * sizeof(float);     // 204,800,000 (mult of 256)
    char* ws = (char*)d_ws;
    float* h    = (float*)(ws);
    float* msum = (float*)(ws + BIG);
    char*  tail = ws + 2 * BIG;
    float* deg  = (float*)tail;                 tail += ((size_t)NNODES * 4 + 255) & ~(size_t)255;
    float* csum = (float*)tail;                 tail += 2048;
    float* css  = (float*)tail;                 tail += 2048;
    float* gsum = (float*)tail;                 tail += (size_t)NB * HD * 4;
    float* gcnt = (float*)tail;                 tail += (size_t)NB * 4;
    // FC buffers reuse the msum region (free after GIN layer 2)
    float* Xcat = msum;                                         // 10240*1536*4 = 62,914,560 B
    float* Pb   = Xcat + (size_t)MROWS * FCIN;                  // 10240*256
    float* Ha   = Pb   + (size_t)MROWS * FCH;
    float* Hb   = Ha   + (size_t)MROWS * FCH;

    float* preds = (float*)d_out;               // [10240]
    float* H3out = (float*)d_out + MROWS;       // [10240 x 256]

    const int edgeWaveBlocks = (NEDGES * 32 + 255) / 256;       // wave per edge
    const int poolWaveBlocks = (NNODES * 32 + 255) / 256;       // wave per node
    const int ginStatBlocks  = (NNODES + GIN_RPB - 1) / GIN_RPB;

    // ---------------- GIN stack ----------------
    hipMemsetAsync(deg,  0, (size_t)NNODES * 4, stream);
    hipMemsetAsync(gsum, 0, (size_t)NB * HD * 4, stream);
    hipMemsetAsync(gcnt, 0, (size_t)NB * 4, stream);

    k_init_gemm<<<NNODES, HD, 0, stream>>>(node_h, w_init, b_init, h);
    k_deg<<<(NEDGES + 255) / 256, 256, 0, stream>>>(dst, deg);

    // layer 1
    hipMemsetAsync(msum, 0, BIG, stream);
    hipMemsetAsync(csum, 0, 2048, stream);
    hipMemsetAsync(css,  0, 2048, stream);
    k_edge_scatter<<<edgeWaveBlocks, 256, 0, stream>>>(h, src, dst, msum);
    k_gin_combine<<<ginStatBlocks, HD, 0, stream>>>(h, msum, deg, csum, css);
    {
        size_t total = (size_t)NNODES * HD;
        k_bn_relu<<<(unsigned)((total + 255) / 256), 256, 0, stream>>>(
            msum, csum, css, g1, be1, h, NNODES, HD);
    }
    // layer 2
    hipMemsetAsync(msum, 0, BIG, stream);
    hipMemsetAsync(csum, 0, 2048, stream);
    hipMemsetAsync(css,  0, 2048, stream);
    k_edge_scatter<<<edgeWaveBlocks, 256, 0, stream>>>(h, src, dst, msum);
    k_gin_combine<<<ginStatBlocks, HD, 0, stream>>>(h, msum, deg, csum, css);
    {
        size_t total = (size_t)NNODES * HD;
        k_bn_relu<<<(unsigned)((total + 255) / 256), 256, 0, stream>>>(
            msum, csum, css, g2, be2, h, NNODES, HD);
    }

    // ---------------- pooling + concat ----------------
    k_pool<<<poolWaveBlocks, 256, 0, stream>>>(h, ngraph, gsum, gcnt);
    {
        size_t total = (size_t)MROWS * FCIN;
        k_concat<<<(unsigned)((total + 255) / 256), 256, 0, stream>>>(
            gsum, gcnt, pg_emb, neigh, Xcat);
    }

    // ---------------- MLP head (WMMA bf16) ----------------
    const int rpb = (MROWS + 127) / 128;  // rows per stats block -> 128 blocks
    dim3 gemmBlk(128);
    dim3 g1d((MROWS + 63) / 64, FCH / 16);
    size_t fcTotal = (size_t)MROWS * FCH;
    unsigned fcGrid = (unsigned)((fcTotal + 255) / 256);

    // FC1: [10240,1536] -> [10240,256]
    k_gemm_bias_wmma<<<g1d, gemmBlk, 0, stream>>>(Xcat, w_fc, b_fc, Pb, MROWS, FCIN, FCH);
    hipMemsetAsync(csum, 0, 2048, stream);
    hipMemsetAsync(css,  0, 2048, stream);
    k_col_stats<<<(MROWS + rpb - 1) / rpb, FCH, 0, stream>>>(Pb, csum, css, MROWS, rpb);
    k_bn_relu<<<fcGrid, 256, 0, stream>>>(Pb, csum, css, gb, bb, Ha, MROWS, FCH);

    // FC2
    k_gemm_bias_wmma<<<g1d, gemmBlk, 0, stream>>>(Ha, w_fc2, b_fc2, Pb, MROWS, FCH, FCH);
    hipMemsetAsync(csum, 0, 2048, stream);
    hipMemsetAsync(css,  0, 2048, stream);
    k_col_stats<<<(MROWS + rpb - 1) / rpb, FCH, 0, stream>>>(Pb, csum, css, MROWS, rpb);
    k_bn_relu<<<fcGrid, 256, 0, stream>>>(Pb, csum, css, gb2, bb2, Hb, MROWS, FCH);

    // FC3 -> H3 goes straight to d_out tail
    k_gemm_bias_wmma<<<g1d, gemmBlk, 0, stream>>>(Hb, w_fc3, b_fc3, Pb, MROWS, FCH, FCH);
    hipMemsetAsync(csum, 0, 2048, stream);
    hipMemsetAsync(css,  0, 2048, stream);
    k_col_stats<<<(MROWS + rpb - 1) / rpb, FCH, 0, stream>>>(Pb, csum, css, MROWS, rpb);
    k_bn_relu<<<fcGrid, 256, 0, stream>>>(Pb, csum, css, gb3, bb3, H3out, MROWS, FCH);

    // FC4 + sigmoid -> preds
    k_fc4_sigmoid<<<(MROWS + 255) / 256, 256, 0, stream>>>(H3out, w_fc4, b_fc4, preds);
}
